// TransBlock_68178310856732
// MI455X (gfx1250) — compile-verified, matrix operand order checked by reading, not run on hardware
//
#include <hip/hip_runtime.h>
#include <hip/hip_bf16.h>

// ---------------------------------------------------------------------------
// Transformer encoder block (pre-norm), MI455X / gfx1250, wave32 + WMMA f16,
// TDM tensor_load_to_lds for GEMM A tiles, async global->LDS for attention K.
// ---------------------------------------------------------------------------

typedef _Float16 v16h __attribute__((ext_vector_type(16)));
typedef float    v8f  __attribute__((ext_vector_type(8)));
typedef _Float16 h8   __attribute__((ext_vector_type(8)));
typedef unsigned int u32x4 __attribute__((ext_vector_type(4)));
typedef int          i32x4 __attribute__((ext_vector_type(4)));
typedef int          i32x8 __attribute__((ext_vector_type(8)));

constexpr int kB   = 2;
constexpr int kL   = 2048;
constexpr int kD   = 1280;
constexpr int kH   = 20;
constexpr int kDH  = 64;
constexpr int kDFF = 5120;
constexpr int kM   = kB * kL;   // 4096 rows

#if defined(__has_builtin)
#  if __has_builtin(__builtin_amdgcn_tensor_load_to_lds)
#    define USE_TDM 1
#  endif
#endif
#ifndef USE_TDM
#  define USE_TDM 0
#endif

// LDS byte offset of a generic pointer into shared memory
static __device__ inline unsigned lds_off(const void* p) {
  return (unsigned)(unsigned long long)(__attribute__((address_space(3))) const void*)p;
}

// Per-lane async 16-byte copy global -> LDS (tracked by ASYNCcnt, no VGPR data)
static __device__ inline void async_copy_b128(const void* gsrc, void* ldst) {
  unsigned l = lds_off(ldst);
  unsigned long long g = (unsigned long long)gsrc;
  asm volatile("global_load_async_to_lds_b128 %0, %1, off"
               :: "v"(l), "v"(g) : "memory");
}
static __device__ inline void wait_async0() {
  asm volatile("s_wait_asynccnt 0x0" ::: "memory");
}

#if USE_TDM
// TDM: load a 2-D f16 tile (rows x rowElems) from a row-major tensor with
// row stride strideElems into LDS at ldsOff. One issue per wave; caller
// must s_wait_tensorcnt then barrier before consumers read LDS.
static __device__ inline void tdm_load_2d_f16(const void* gaddr, unsigned ldsOff,
                                              unsigned rowElems, unsigned rows,
                                              unsigned tensorW, unsigned tensorH,
                                              unsigned strideElems) {
  unsigned long long ga = (unsigned long long)gaddr;
  u32x4 g0;
  g0[0] = 1u;                                           // count=1 valid D#
  g0[1] = ldsOff;                                       // lds_addr
  g0[2] = (unsigned)ga;                                 // global_addr[31:0]
  g0[3] = (unsigned)((ga >> 32) & 0x1FFFFFFu)           // global_addr[56:32]
        | (2u << 30);                                   // type = 2 ("image")
  i32x8 g1;
  g1[0] = (int)(1u << 16);                              // data_size=1 (2 bytes)
  g1[1] = (int)(tensorW << 16);                         // tensor_dim0[15:0]
  g1[2] = (int)((tensorW >> 16) | (tensorH << 16));     // dim0 hi | dim1 lo
  g1[3] = (int)((tensorH >> 16) | (rowElems << 16));    // dim1 hi | tile_dim0
  g1[4] = (int)rows;                                    // tile_dim1 (tile_dim2=0)
  g1[5] = (int)strideElems;                             // tensor_dim0_stride lo
  g1[6] = 0;                                            // stride hi / dim1 stride
  g1[7] = 0;
  i32x4 z4 = {0, 0, 0, 0};
#if defined(__clang_major__) && __clang_major__ >= 23
  i32x8 z8 = {0, 0, 0, 0, 0, 0, 0, 0};
  __builtin_amdgcn_tensor_load_to_lds(g0, g1, z4, z4, z8, 0);
#else
  __builtin_amdgcn_tensor_load_to_lds(g0, g1, z4, z4, 0);
#endif
}
#endif  // USE_TDM

// ---------------------------------------------------------------------------
// f32 -> f16 conversion (weights, done per call into workspace)
// ---------------------------------------------------------------------------
__global__ __launch_bounds__(256)
void cvt_f32_to_f16_kernel(const float* __restrict__ src,
                           _Float16* __restrict__ dst, int n) {
  for (int i = blockIdx.x * blockDim.x + threadIdx.x; i < n;
       i += gridDim.x * blockDim.x) {
    dst[i] = (_Float16)src[i];
  }
}

// ---------------------------------------------------------------------------
// LayerNorm: one block (256 threads = 8 waves) per row of D=1280, f16 output.
// ---------------------------------------------------------------------------
__global__ __launch_bounds__(256)
void layernorm_f16_kernel(const float* __restrict__ x,
                          const float* __restrict__ gamma,
                          const float* __restrict__ beta,
                          _Float16* __restrict__ out) {
  __shared__ float rs[8], rs2[8];
  const int row = blockIdx.x;
  const float* xr = x + (size_t)row * kD;
  float s = 0.f, s2 = 0.f;
  for (int i = threadIdx.x; i < kD; i += 256) {
    float v = xr[i];
    s += v; s2 += v * v;
  }
  #pragma unroll
  for (int off = 1; off < 32; off <<= 1) {
    s  += __shfl_xor(s,  off, 32);
    s2 += __shfl_xor(s2, off, 32);
  }
  const int lane = threadIdx.x & 31, wave = threadIdx.x >> 5;
  if (lane == 0) { rs[wave] = s; rs2[wave] = s2; }
  __syncthreads();
  if (wave == 0) {
    s  = (lane < 8) ? rs[lane]  : 0.f;
    s2 = (lane < 8) ? rs2[lane] : 0.f;
    #pragma unroll
    for (int off = 1; off < 8; off <<= 1) {
      s  += __shfl_xor(s,  off, 32);
      s2 += __shfl_xor(s2, off, 32);
    }
    if (lane == 0) { rs[0] = s; rs2[0] = s2; }
  }
  __syncthreads();
  const float mean = rs[0] * (1.f / kD);
  const float var  = rs2[0] * (1.f / kD) - mean * mean;
  const float inv  = rsqrtf(var + 1e-5f);
  for (int i = threadIdx.x; i < kD; i += 256) {
    out[(size_t)row * kD + i] =
        (_Float16)((xr[i] - mean) * inv * gamma[i] + beta[i]);
  }
}

// ---------------------------------------------------------------------------
// Tiled WMMA GEMM: C[M,N] = A[M,K](f16) * B[K,N](f16) then
// optional *scale, +bias(f32), relu, +residual(f32); writes f32 and/or f16.
// Block = 256 threads (8 waves, 4x2 wave grid), block tile 128x64, K-step 32.
// A tile moved by the Tensor Data Mover; B tile transposed through registers.
// ---------------------------------------------------------------------------
__global__ __launch_bounds__(256)
void gemm_f16_wmma_kernel(const _Float16* __restrict__ A,
                          const _Float16* __restrict__ Bm,
                          float* __restrict__ Cf,
                          _Float16* __restrict__ Ch,
                          const float* __restrict__ bias,
                          const float* __restrict__ resid,
                          int Mdim, int Ndim, int Kdim,
                          float scale, int relu) {
  __shared__ alignas(32) _Float16 As[128][32];   // [m][k]
  __shared__ alignas(32) _Float16 Bt[64][32];    // transposed: [n][k]

  const int tid  = threadIdx.x;
  const int lane = tid & 31;
  const int wave = tid >> 5;           // 0..7
  const int wm   = wave >> 1;          // 0..3 along M
  const int wn   = wave & 1;           // 0..1 along N
  const int laneRow = lane & 15;
  const int laneHi  = lane >> 4;       // 0 or 1
  const int rowBase = blockIdx.y * 128;
  const int colBase = blockIdx.x * 64;

  v8f acc[2][2] = {};

  for (int k0 = 0; k0 < Kdim; k0 += 32) {
    // --- stage A tile 128x32 (row-major) ---
#if USE_TDM
    if (tid < 32) {   // wave 0 drives the Tensor Data Mover
      tdm_load_2d_f16(A + (size_t)rowBase * Kdim + k0, lds_off(&As[0][0]),
                      /*rowElems=*/32, /*rows=*/128,
                      (unsigned)Kdim, (unsigned)Mdim, (unsigned)Kdim);
      __builtin_amdgcn_s_wait_tensorcnt((short)0);
    }
#else
    {
      const int r  = tid >> 1;
      const int ch = (tid & 1) * 16;
      const _Float16* src = A + (size_t)(rowBase + r) * Kdim + k0 + ch;
      *(h8*)&As[r][ch]     = *(const h8*)src;
      *(h8*)&As[r][ch + 8] = *(const h8*)(src + 8);
    }
#endif
    // --- stage B tile 32x64, transposed into Bt[n][k] ---
    {
      const int kk = tid >> 3;            // 0..31
      const int c8 = (tid & 7) * 8;       // 0..56
      const _Float16* src = Bm + (size_t)(k0 + kk) * Ndim + colBase + c8;
      h8 bv = *(const h8*)src;
      #pragma unroll
      for (int j = 0; j < 8; ++j) Bt[c8 + j][kk] = bv[j];
    }
    __syncthreads();

    // speculative prefetch of the next A tile (global_prefetch_b8)
    if (k0 + 32 < Kdim) {
      __builtin_prefetch(A + (size_t)(rowBase + (tid >> 1)) * Kdim + k0 + 32,
                         0, 1);
    }

    #pragma unroll
    for (int mt = 0; mt < 2; ++mt) {
      v16h afrag = *(const v16h*)&As[wm * 32 + mt * 16 + laneRow][laneHi * 16];
      #pragma unroll
      for (int nt = 0; nt < 2; ++nt) {
        v16h bfrag = *(const v16h*)&Bt[wn * 32 + nt * 16 + laneRow][laneHi * 16];
        acc[mt][nt] = __builtin_amdgcn_wmma_f32_16x16x32_f16(
            false, afrag, false, bfrag, (short)0, acc[mt][nt], false, false);
      }
    }
    __syncthreads();
  }

  // --- epilogue (C layout: lanes 0-15 -> rows 0..7, lanes 16-31 -> 8..15) ---
  #pragma unroll
  for (int mt = 0; mt < 2; ++mt) {
    #pragma unroll
    for (int nt = 0; nt < 2; ++nt) {
      #pragma unroll
      for (int r = 0; r < 8; ++r) {
        const int row = rowBase + wm * 32 + mt * 16 + laneHi * 8 + r;
        const int col = colBase + wn * 32 + nt * 16 + laneRow;
        float v = acc[mt][nt][r] * scale;
        if (bias)  v += bias[col];
        if (relu)  v = fmaxf(v, 0.f);
        const size_t idx = (size_t)row * Ndim + col;
        if (resid) v += resid[idx];
        if (Cf) Cf[idx] = v;
        if (Ch) Ch[idx] = (_Float16)v;
      }
    }
  }
}

// ---------------------------------------------------------------------------
// Flash-style attention, one head-slice per block.y/z, 8 q-tiles per block.
// Each wave owns a 16-row Q tile; 32-key K/V tiles staged in LDS per step
// (K via async global->LDS copies on ASYNCcnt, V transposed via registers).
// QK^T: 4 WMMA; online softmax in the C-fragment row layout; PV: 4 WMMA.
// q is pre-scaled by 1/sqrt(DH) at the QKV GEMM.
// ---------------------------------------------------------------------------
__global__ __launch_bounds__(256)
void attn_wmma_kernel(const _Float16* __restrict__ Q,
                      const _Float16* __restrict__ Km,
                      const _Float16* __restrict__ Vm,
                      _Float16* __restrict__ O) {
  __shared__ alignas(32) _Float16 Ks[32][64];      // [key][dim]
  __shared__ alignas(32) _Float16 VtT[64][32];     // transposed: [dim][key]
  __shared__ alignas(32) _Float16 Ps[8][16][32];   // per-wave P staging

  const int tid  = threadIdx.x;
  const int lane = tid & 31;
  const int wave = tid >> 5;
  const int laneRow = lane & 15;
  const int laneHi  = lane >> 4;
  const int h  = blockIdx.y;
  const int b  = blockIdx.z;
  const int q0 = blockIdx.x * 128 + wave * 16;

  // Q fragments (loop invariant): kd chunks [0,32) and [32,64)
  v16h aq[2];
  {
    const _Float16* qp =
        Q + ((size_t)(b * kL + q0 + laneRow) * kD) + h * kDH + laneHi * 16;
    aq[0] = *(const v16h*)qp;
    aq[1] = *(const v16h*)(qp + 32);
  }

  float mrow[8], lrow[8];
  v8f o[4] = {};
  #pragma unroll
  for (int r = 0; r < 8; ++r) { mrow[r] = -1e30f; lrow[r] = 0.f; }

  for (int kb = 0; kb < kL; kb += 32) {
    // cooperative stage: K async (memory->LDS, ASYNCcnt), V transposed
    {
      const int kk = tid >> 3;        // 0..31 key
      const int c8 = (tid & 7) * 8;   // 0..56 dim
      const size_t base = ((size_t)(b * kL + kb + kk) * kD) + h * kDH + c8;
      async_copy_b128(Km + base, &Ks[kk][c8]);
      h8 vv = *(const h8*)(Vm + base);
      #pragma unroll
      for (int j = 0; j < 8; ++j) VtT[c8 + j][kk] = vv[j];
      wait_async0();
    }
    __syncthreads();

    // scores: two 16x16 tiles (keys kb..+15, kb+16..+31), K-depth 64 = 2x32
    v8f s[2] = {};
    #pragma unroll
    for (int t = 0; t < 2; ++t) {
      #pragma unroll
      for (int kc = 0; kc < 2; ++kc) {
        v16h bfr = *(const v16h*)&Ks[t * 16 + laneRow][kc * 32 + laneHi * 16];
        s[t] = __builtin_amdgcn_wmma_f32_16x16x32_f16(
            false, aq[kc], false, bfr, (short)0, s[t], false, false);
      }
    }

    // online softmax across this 16x32 tile (rows live across 16-lane halves)
    float p0[8], p1[8];
    #pragma unroll
    for (int r = 0; r < 8; ++r) {
      float mx = fmaxf(s[0][r], s[1][r]);
      #pragma unroll
      for (int off = 1; off < 16; off <<= 1)
        mx = fmaxf(mx, __shfl_xor(mx, off, 32));
      const float mnew = fmaxf(mrow[r], mx);
      const float sc   = __expf(mrow[r] - mnew);
      p0[r] = __expf(s[0][r] - mnew);
      p1[r] = __expf(s[1][r] - mnew);
      float rsum = p0[r] + p1[r];
      #pragma unroll
      for (int off = 1; off < 16; off <<= 1)
        rsum += __shfl_xor(rsum, off, 32);
      lrow[r] = lrow[r] * sc + rsum;
      mrow[r] = mnew;
      #pragma unroll
      for (int c = 0; c < 4; ++c) o[c][r] *= sc;
    }

    // stage P through this wave's private LDS slice to get A-frag layout
    #pragma unroll
    for (int r = 0; r < 8; ++r) {
      Ps[wave][laneHi * 8 + r][laneRow]      = (_Float16)p0[r];
      Ps[wave][laneHi * 8 + r][16 + laneRow] = (_Float16)p1[r];
    }
    // in-wave LDS RAW: compiler inserts s_wait_dscnt; no block barrier needed
    v16h ap = *(const v16h*)&Ps[wave][laneRow][laneHi * 16];

    // PV: O[16x64] += P[16x32] x V[32x64]
    #pragma unroll
    for (int c = 0; c < 4; ++c) {
      v16h bv = *(const v16h*)&VtT[c * 16 + laneRow][laneHi * 16];
      o[c] = __builtin_amdgcn_wmma_f32_16x16x32_f16(
          false, ap, false, bv, (short)0, o[c], false, false);
    }
    __syncthreads();
  }

  // final normalize + store (f16), layout (B,L,H,DH) == row-major (M, D)
  #pragma unroll
  for (int c = 0; c < 4; ++c) {
    #pragma unroll
    for (int r = 0; r < 8; ++r) {
      const int row = q0 + laneHi * 8 + r;
      const float val = o[c][r] / lrow[r];
      O[((size_t)(b * kL + row)) * kD + h * kDH + c * 16 + laneRow] =
          (_Float16)val;
    }
  }
}

// ---------------------------------------------------------------------------
// Launcher
// ---------------------------------------------------------------------------
static inline size_t align256(size_t v) { return (v + 255) & ~(size_t)255; }

extern "C" void kernel_launch(void* const* d_in, const int* in_sizes, int n_in,
                              void* d_out, int out_size, void* d_ws,
                              size_t ws_size, hipStream_t stream) {
  const float* x   = (const float*)d_in[0];
  // d_in[1] = y (unused by the module)
  const float* Wq  = (const float*)d_in[2];
  const float* Wk  = (const float*)d_in[3];
  const float* Wv  = (const float*)d_in[4];
  const float* Wo  = (const float*)d_in[5];
  const float* bo  = (const float*)d_in[6];
  const float* W1  = (const float*)d_in[7];
  const float* b1  = (const float*)d_in[8];
  const float* W2  = (const float*)d_in[9];
  const float* b2  = (const float*)d_in[10];
  const float* g1  = (const float*)d_in[11];
  const float* be1 = (const float*)d_in[12];
  const float* g2  = (const float*)d_in[13];
  const float* be2 = (const float*)d_in[14];

  const size_t nDD = (size_t)kD * kD;
  const size_t nDF = (size_t)kD * kDFF;
  const size_t nMD = (size_t)kM * kD;
  const size_t nMF = (size_t)kM * kDFF;

  char* p = (char*)d_ws;
  auto takeH = [&](size_t n) { _Float16* r = (_Float16*)p; p += align256(n * 2); return r; };
  auto takeF = [&](size_t n) { float* r = (float*)p;      p += align256(n * 4); return r; };

  _Float16* wq16 = takeH(nDD);
  _Float16* wk16 = takeH(nDD);
  _Float16* wv16 = takeH(nDD);
  _Float16* wo16 = takeH(nDD);
  _Float16* w1h  = takeH(nDF);
  _Float16* w2h  = takeH(nDF);
  _Float16* xn1  = takeH(nMD);
  _Float16* q16  = takeH(nMD);
  _Float16* k16  = takeH(nMD);
  _Float16* v16  = takeH(nMD);
  _Float16* ao16 = takeH(nMD);
  _Float16* xn2  = takeH(nMD);
  _Float16* h16  = takeH(nMF);
  float*    x1f  = takeF(nMD);

  // weight conversions (f32 -> f16)
  auto cvt = [&](const float* s, _Float16* d, size_t n) {
    int blocks = (int)((n + 256 * 4 - 1) / (256 * 4));
    cvt_f32_to_f16_kernel<<<blocks, 256, 0, stream>>>(s, d, (int)n);
  };
  cvt(Wq, wq16, nDD);
  cvt(Wk, wk16, nDD);
  cvt(Wv, wv16, nDD);
  cvt(Wo, wo16, nDD);
  cvt(W1, w1h, nDF);
  cvt(W2, w2h, nDF);

  // --- pre-norm attention ---
  layernorm_f16_kernel<<<kM, 256, 0, stream>>>(x, g1, be1, xn1);

  dim3 gqkv(kD / 64, kM / 128);
  // q scaled by 1/sqrt(DH) = 0.125
  gemm_f16_wmma_kernel<<<gqkv, 256, 0, stream>>>(
      xn1, wq16, nullptr, q16, nullptr, nullptr, kM, kD, kD, 0.125f, 0);
  gemm_f16_wmma_kernel<<<gqkv, 256, 0, stream>>>(
      xn1, wk16, nullptr, k16, nullptr, nullptr, kM, kD, kD, 1.0f, 0);
  gemm_f16_wmma_kernel<<<gqkv, 256, 0, stream>>>(
      xn1, wv16, nullptr, v16, nullptr, nullptr, kM, kD, kD, 1.0f, 0);

  dim3 gattn(kL / 128, kH, kB);
  attn_wmma_kernel<<<gattn, 256, 0, stream>>>(q16, k16, v16, ao16);

  // x1 = x + ao @ Wo + bo   (f32)
  gemm_f16_wmma_kernel<<<gqkv, 256, 0, stream>>>(
      ao16, wo16, x1f, nullptr, bo, x, kM, kD, kD, 1.0f, 0);

  // --- pre-norm FFN ---
  layernorm_f16_kernel<<<kM, 256, 0, stream>>>(x1f, g2, be2, xn2);

  dim3 gff1(kDFF / 64, kM / 128);
  gemm_f16_wmma_kernel<<<gff1, 256, 0, stream>>>(
      xn2, w1h, nullptr, h16, b1, nullptr, kM, kDFF, kD, 1.0f, 1);

  dim3 gff2(kD / 64, kM / 128);
  gemm_f16_wmma_kernel<<<gff2, 256, 0, stream>>>(
      h16, w2h, (float*)d_out, nullptr, b2, x1f, kM, kD, kDFF, 1.0f, 0);
}
